// RNNClassifier_6871947674038
// MI455X (gfx1250) — compile-verified, matrix operand order checked by reading, not run on hardware
//
#include <hip/hip_runtime.h>
#include <hip/hip_bf16.h>
#include <math.h>

// ---------------------------------------------------------------------------
// RNNClassifier on MI455X (gfx1250, wave32, WMMA)
//
// One wave = one 16-batch tile. Per timestep:
//   acc(16x32) = x_t(16x32) @ W_ih^T + bias   -> 2x v_wmma_f32_16x16x32_f16
//   acc       += h(16x32)   @ W_hh^T          -> 2x v_wmma_f32_16x16x32_f16 (chain)
//   h = tanh(acc)                             -> v_tanh_f32
//   D-layout -> A-layout via per-wave LDS tile (s_wait_dscnt 0)
// Next timestep's input fragment is loaded right after the projection WMMAs
// (software pipelining), so global latency overlaps the recurrence chain.
// Layers are template-specialized: no runtime branches in the hot loop.
// ---------------------------------------------------------------------------

#define BB 512
#define TT 2048
#define HH 32

typedef __attribute__((ext_vector_type(16))) _Float16 v16h;
typedef __attribute__((ext_vector_type(8)))  _Float16 v8h;
typedef __attribute__((ext_vector_type(8)))  float    v8f;
typedef __attribute__((ext_vector_type(4)))  float    v4f;

#if __has_builtin(__builtin_amdgcn_tanhf)
#define TANHF(x) __builtin_amdgcn_tanhf(x)
#else
#define TANHF(x) tanhf(x)
#endif

// A/B fragment K index for half i (0..15) given lane group g (0/1).
__device__ __forceinline__ constexpr int frag_k(int i, int g) {
    return (i < 8) ? (8 * g + i) : (16 + 8 * g + (i - 8));
}

// Load the 16x32 input A-fragment for timestep t.
template <bool FIRST>
__device__ __forceinline__ v16h load_x_frag(const void* __restrict__ seq_in_v,
                                            int b0, int m, int g, int t) {
    v16h xa;
    if (FIRST) {
        // x is f32 [B, T, 6]; pad K=6..31 with zeros, branchlessly.
        const float* xp = (const float*)seq_in_v + ((size_t)(b0 + m) * TT + t) * 6;
        float xv[6];
#pragma unroll
        for (int i = 0; i < 6; ++i) xv[i] = xp[i];   // unconditional loads
#pragma unroll
        for (int i = 0; i < 16; ++i) {
            // k = frag_k(i,g) < 6 only possible when g==0 and i<6 -> cndmask, no branch
            _Float16 v = (i < 6) ? (_Float16)xv[i < 6 ? i : 0] : (_Float16)0.0f;
            xa[i] = (g == 0 && i < 6) ? v : (_Float16)0.0f;
        }
    } else {
        const _Float16* p = (const _Float16*)seq_in_v
                          + ((size_t)(b0 + m) * TT + t) * 32 + 8 * g;
        v8h lo = *(const v8h*)p;         // K = 8g .. 8g+7
        v8h hi = *(const v8h*)(p + 16);  // K = 16+8g .. 16+8g+7
#pragma unroll
        for (int i = 0; i < 8; ++i) { xa[i] = lo[i]; xa[8 + i] = hi[i]; }
    }
    return xa;
}

template <bool FIRST, bool WRITE_SEQ, bool WRITE_H>
__global__ __launch_bounds__(32)
void rnn_layer_kernel(const void* __restrict__ seq_in_v,   // f32 [B,T,6] if FIRST else f16 [B,T,32]
                      _Float16*   __restrict__ seq_out,    // f16 [B,T,32]
                      const float* __restrict__ W_ih,      // [32, d_in] row-major
                      const float* __restrict__ W_hh,      // [32, 32] row-major
                      const float* __restrict__ b_ih,      // [32]
                      const float* __restrict__ b_hh,      // [32]
                      float*      __restrict__ h_last)     // f32 [B,32]
{
    __shared__ float hbuf[16 * 32];   // per-wave h tile, [row m][col k], f32

    const int lane = threadIdx.x & 31;
    const int m    = lane & 15;       // A row / B col / C col
    const int g    = lane >> 4;      // lane group
    const int b0   = blockIdx.x * 16; // batch tile base

    // ---- Weight B-fragments: B[k][n] = W[n][k], two N tiles, f16 ----
    v16h Bih[2], Bhh[2];
#pragma unroll
    for (int nt = 0; nt < 2; ++nt) {
        const int n = m + 16 * nt;
        v16h bi, bh;
#pragma unroll
        for (int i = 0; i < 16; ++i) {
            const int k = frag_k(i, g);
            float wih;
            if (FIRST) {
                // unconditional read at a clamped index, then branchless select
                float w = W_ih[n * 6 + (k < 6 ? k : 0)];
                wih = (k < 6) ? w : 0.0f;
            } else {
                wih = W_ih[n * 32 + k];
            }
            bi[i] = (_Float16)wih;
            bh[i] = (_Float16)W_hh[n * 32 + k];
        }
        Bih[nt] = bi;
        Bhh[nt] = bh;
    }

    // ---- Bias as C fragments (broadcast down rows) ----
    v8f cb[2];
#pragma unroll
    for (int nt = 0; nt < 2; ++nt) {
        const float bsum = b_ih[m + 16 * nt] + b_hh[m + 16 * nt];
#pragma unroll
        for (int r = 0; r < 8; ++r) cb[nt][r] = bsum;
    }

    // h0 = 0 (A-fragment layout)
    v16h ha;
#pragma unroll
    for (int i = 0; i < 16; ++i) ha[i] = (_Float16)0.0f;

    // Prime the pipeline with x_0.
    v16h xa = load_x_frag<FIRST>(seq_in_v, b0, m, g, 0);

    for (int t = 0; t < TT; ++t) {
        // ---- input projection (off the recurrence chain) ----
        v8f acc0 = __builtin_amdgcn_wmma_f32_16x16x32_f16(false, xa, false, Bih[0],
                                                          (short)0, cb[0], false, false);
        v8f acc1 = __builtin_amdgcn_wmma_f32_16x16x32_f16(false, xa, false, Bih[1],
                                                          (short)0, cb[1], false, false);

        // ---- issue next timestep's input load; overlaps the chain below ----
        const int tn = (t + 1 < TT) ? (t + 1) : (TT - 1);   // clamped, branchless
        v16h xn = load_x_frag<FIRST>(seq_in_v, b0, m, g, tn);

        // ---- recurrence: acc += h @ W_hh^T (dependency chain) ----
        acc0 = __builtin_amdgcn_wmma_f32_16x16x32_f16(false, ha, false, Bhh[0],
                                                      (short)0, acc0, false, false);
        acc1 = __builtin_amdgcn_wmma_f32_16x16x32_f16(false, ha, false, Bhh[1],
                                                      (short)0, acc1, false, false);

        // ---- tanh + scatter D-layout into LDS [m][k] ----
#pragma unroll
        for (int r = 0; r < 8; ++r) {
            acc0[r] = TANHF(acc0[r]);
            acc1[r] = TANHF(acc1[r]);
            hbuf[(r + 8 * g) * 32 + m]      = acc0[r];
            hbuf[(r + 8 * g) * 32 + m + 16] = acc1[r];
        }
        asm volatile("s_wait_dscnt 0" ::: "memory");

        // ---- reload h as f16 A-fragment ----
        {
            const float* hp = &hbuf[m * 32 + 8 * g];
            v4f f0 = *(const v4f*)hp;          // K = 8g..8g+3
            v4f f1 = *(const v4f*)(hp + 4);    // K = 8g+4..8g+7
            v4f f2 = *(const v4f*)(hp + 16);   // K = 16+8g..+3
            v4f f3 = *(const v4f*)(hp + 20);   // K = 16+8g+4..+7
#pragma unroll
            for (int i = 0; i < 4; ++i) {
                ha[i]      = (_Float16)f0[i];
                ha[4 + i]  = (_Float16)f1[i];
                ha[8 + i]  = (_Float16)f2[i];
                ha[12 + i] = (_Float16)f3[i];
            }
        }

        // ---- emit sequence output (f16, A-friendly row-major layout) ----
        if (WRITE_SEQ) {
            _Float16* q = seq_out + ((size_t)(b0 + m) * TT + t) * 32 + 8 * g;
            v8h lo, hi;
#pragma unroll
            for (int i = 0; i < 8; ++i) { lo[i] = ha[i]; hi[i] = ha[8 + i]; }
            *(v8h*)q        = lo;
            *(v8h*)(q + 16) = hi;
        }

        xa = xn;
    }

    // ---- final hidden state in f32 (from LDS, exact post-tanh values) ----
    if (WRITE_H) {
        const float* hp = &hbuf[m * 32 + 8 * g];
        float* q = h_last + (size_t)(b0 + m) * 32 + 8 * g;
        *(v4f*)q        = *(const v4f*)hp;
        *(v4f*)(q + 4)  = *(const v4f*)(hp + 4);
        *(v4f*)(q + 16) = *(const v4f*)(hp + 16);
        *(v4f*)(q + 20) = *(const v4f*)(hp + 20);
    }
}

// out[b,c] = h_last[b,:] . fc_w[c,:] + fc_b[c]   (512 x 6, trivial)
__global__ __launch_bounds__(256)
void fc_head_kernel(const float* __restrict__ h, const float* __restrict__ fc_w,
                    const float* __restrict__ fc_b, float* __restrict__ out)
{
    int idx = blockIdx.x * blockDim.x + threadIdx.x;
    if (idx >= BB * 6) return;
    int b = idx / 6, c = idx % 6;
    float s = fc_b[c];
#pragma unroll
    for (int k = 0; k < HH; ++k) s += h[b * HH + k] * fc_w[c * HH + k];
    out[idx] = s;
}

extern "C" void kernel_launch(void* const* d_in, const int* in_sizes, int n_in,
                              void* d_out, int out_size, void* d_ws, size_t ws_size,
                              hipStream_t stream) {
    (void)in_sizes; (void)n_in; (void)out_size; (void)ws_size;

    const float* x         = (const float*)d_in[0];  // [512,2048,6]
    const float* W_ih0     = (const float*)d_in[1];  // [32,6]
    const float* W_ih_rest = (const float*)d_in[2];  // [2,32,32]
    const float* W_hh      = (const float*)d_in[3];  // [3,32,32]
    const float* b_ih      = (const float*)d_in[4];  // [3,32]
    const float* b_hh      = (const float*)d_in[5];  // [3,32]
    const float* fc_w      = (const float*)d_in[6];  // [6,32]
    const float* fc_b      = (const float*)d_in[7];  // [6]
    float* out = (float*)d_out;                      // [512,6]

    // workspace: seq f16 [B,T,32] (64 MiB) + h_last f32 [B,32] (64 KiB)
    char* ws = (char*)d_ws;
    _Float16* seq   = (_Float16*)ws;
    float*    hlast = (float*)(ws + (size_t)BB * TT * HH * sizeof(_Float16));

    // Layer 0: reads raw x (D=6), writes seq
    rnn_layer_kernel<true, true, false><<<BB / 16, 32, 0, stream>>>(
        x, seq, W_ih0, W_hh + 0 * HH * HH, b_ih + 0 * HH, b_hh + 0 * HH, nullptr);
    // Layer 1: in-place on seq (read t and t+1 before writing t, same wave)
    rnn_layer_kernel<false, true, false><<<BB / 16, 32, 0, stream>>>(
        seq, seq, W_ih_rest + 0 * HH * HH, W_hh + 1 * HH * HH,
        b_ih + 1 * HH, b_hh + 1 * HH, nullptr);
    // Layer 2: only the final hidden state matters
    rnn_layer_kernel<false, false, true><<<BB / 16, 32, 0, stream>>>(
        seq, nullptr, W_ih_rest + 1 * HH * HH, W_hh + 2 * HH * HH,
        b_ih + 2 * HH, b_hh + 2 * HH, hlast);
    // FC head
    fc_head_kernel<<<(BB * 6 + 255) / 256, 256, 0, stream>>>(hlast, fc_w, fc_b, out);
}
